// OutputPPBlock_3822520894069
// MI455X (gfx1250) — matrix operand order repaired and checked by reading, not compiled, not run on hardware
//
#include <hip/hip_runtime.h>
#include <hip/hip_bf16.h>

typedef __attribute__((ext_vector_type(16))) _Float16 v16h;
typedef __attribute__((ext_vector_type(8)))  float    v8f;

#define HIDDEN 128
#define NRAD   6

// ---------------------------------------------------------------------------
// Zero the per-node accumulator (must re-init every launch: graph replays).
// ---------------------------------------------------------------------------
__global__ void zero_acc(float* __restrict__ p, int n) {
  int i = (blockIdx.x * blockDim.x + threadIdx.x) * 4;
  if (i < n) {
    *(float4*)(p + i) = make_float4(0.f, 0.f, 0.f, 0.f);
  }
}

// ---------------------------------------------------------------------------
// Convert W1/W2 (f32, [128,128] row-major) to f16 scratch copies.
// ---------------------------------------------------------------------------
__global__ void cvt_w(const float* __restrict__ W1, const float* __restrict__ W2,
                      _Float16* __restrict__ W1h, _Float16* __restrict__ W2h, int n) {
  int i = blockIdx.x * blockDim.x + threadIdx.x;
  if (i < n) {
    W1h[i] = (_Float16)W1[i];
  } else if (i < 2 * n) {
    W2h[i - n] = (_Float16)W2[i - n];
  }
}

// ---------------------------------------------------------------------------
// Edge phase: one wave per edge. Lane owns 4 channels (lane*4 .. lane*4+3).
//   emb[c] = dot(rbf[e], W_rbf[c,:])      (K = 6, coeffs held in registers)
//   val[c] = x[row][c] * emb[c]           (coalesced 512B float4 gather)
//   acc[col][c] += val[c]                 (global_atomic_add_f32, L2-resident)
// ---------------------------------------------------------------------------
__global__ void edge_scatter(const float* __restrict__ x,
                             const float* __restrict__ rbf,
                             const int*   __restrict__ ei,     // [2, E]
                             const float* __restrict__ Wrbf,   // [128, 6]
                             float*       __restrict__ acc,    // [N, 128]
                             int E) {
  const int lane = threadIdx.x & 31;
  const int wid  = (blockIdx.x * blockDim.x + threadIdx.x) >> 5;
  const int nw   = (gridDim.x * blockDim.x) >> 5;
  const int c0   = lane * 4;

  // Per-lane rbf-embedding weights: 4 rows x 6 coeffs (L2/L0 cached, read once)
  float w[4][NRAD];
#pragma unroll
  for (int c = 0; c < 4; ++c)
#pragma unroll
    for (int j = 0; j < NRAD; ++j)
      w[c][j] = Wrbf[(c0 + c) * NRAD + j];

  for (int e = wid; e < E; e += nw) {
    const int row = ei[e];
    const int col = ei[E + e];

    // rbf[e, 0:6] -- 24B, 8B aligned (stride 24B): three float2 loads
    const float2 r01 = *(const float2*)(rbf + (size_t)e * NRAD + 0);
    const float2 r23 = *(const float2*)(rbf + (size_t)e * NRAD + 2);
    const float2 r45 = *(const float2*)(rbf + (size_t)e * NRAD + 4);

    const float4 xv = *(const float4*)(x + (size_t)row * HIDDEN + c0);

    float val[4];
#pragma unroll
    for (int c = 0; c < 4; ++c) {
      float emb = w[c][0] * r01.x + w[c][1] * r01.y +
                  w[c][2] * r23.x + w[c][3] * r23.y +
                  w[c][4] * r45.x + w[c][5] * r45.y;
      float xc = (c == 0) ? xv.x : (c == 1) ? xv.y : (c == 2) ? xv.z : xv.w;
      val[c] = xc * emb;
    }

    unsigned long long base = (unsigned long long)(acc + (size_t)col * HIDDEN + c0);
#pragma unroll
    for (int c = 0; c < 4; ++c) {
      unsigned long long a = base + 4ull * c;
      asm volatile("global_atomic_add_f32 %0, %1, off"
                   :: "v"(a), "v"(val[c]) : "memory");
    }
  }
}

// ---------------------------------------------------------------------------
// Fast SiLU: x * sigmoid(x) using v_exp_f32 + v_rcp_f32 (no IEEE div chain).
// Denominator is in (1, inf): rcp is safe and ~1 ulp.
// ---------------------------------------------------------------------------
__device__ __forceinline__ float fast_silu(float v) {
  return v * __builtin_amdgcn_rcpf(1.0f + __expf(-v));
}

// ---------------------------------------------------------------------------
// MLP phase: one wave computes one 16-node tile. 8 independent waves / block.
// Layer 1: h = silu(acc @ W1^T + b1)   via v_wmma_f32_16x16x32_f16
// Layer 2: out = h @ W2^T + b2         via v_wmma_f32_16x16x32_f16
// h is staged in a wave-private LDS slab as f16 (A-matrix layout source).
// ---------------------------------------------------------------------------
__global__ void mlp(const float*    __restrict__ acc,   // [N,128] f32
                    const _Float16* __restrict__ W1h,   // [128,128] f16
                    const float*    __restrict__ b1,
                    const _Float16* __restrict__ W2h,   // [128,128] f16
                    const float*    __restrict__ b2,
                    float*          __restrict__ out,   // [N,128] f32
                    int ntiles) {
  __shared__ _Float16 hbuf[8 * 16 * HIDDEN];            // 32 KB, 4 KB per wave

  const int lane = threadIdx.x & 31;
  const int wv   = threadIdx.x >> 5;
  const int tile = blockIdx.x * 8 + wv;
  if (tile >= ntiles) return;

  _Float16* hl = hbuf + wv * 16 * HIDDEN;

  const int m     = lane & 15;                 // A row / B,C column index
  const int khalf = (lane < 16) ? 0 : 8;       // unused: A also loads at kb16
  const int kb16  = (lane < 16) ? 0 : 16;      // B-matrix K half
  const int rbase = (lane < 16) ? 0 : 8;       // C-matrix row base

  const float* arow = acc + ((size_t)tile * 16 + m) * HIDDEN;

  // ---- Layer 1: 8 column tiles x (K=128 in 4 steps of 32) ----
  for (int nt = 0; nt < 8; ++nt) {
    const int nb = nt * 16;
    const float bias = b1[nb + m];
    v8f c;
#pragma unroll
    for (int i = 0; i < 8; ++i) c[i] = bias;

#pragma unroll
    for (int k0 = 0; k0 < HIDDEN; k0 += 32) {
      v16h a;
#pragma unroll
      for (int i = 0; i < 8; ++i) {
        a[i]     = (_Float16)arow[k0 + kb16 + i];
        a[8 + i] = (_Float16)arow[k0 + kb16 + 8 + i];
      }
      const v16h b = *(const v16h*)(W1h + (size_t)(nb + m) * HIDDEN + k0 + kb16);
      c = __builtin_amdgcn_wmma_f32_16x16x32_f16(false, a, false, b,
                                                 (short)0, c, false, false);
    }
#pragma unroll
    for (int i = 0; i < 8; ++i) {
      hl[(rbase + i) * HIDDEN + nb + m] = (_Float16)fast_silu(c[i]);
    }
  }

  // Same-wave DS store->load ordering fence (waves are independent; no barrier)
  asm volatile("s_wait_dscnt 0" ::: "memory");

  // ---- Layer 2 ----
  const _Float16* hrow = hl + m * HIDDEN;
  float* orow0 = out + (size_t)tile * 16 * HIDDEN;
  for (int nt = 0; nt < 8; ++nt) {
    const int nb = nt * 16;
    const float bias = b2[nb + m];
    v8f c;
#pragma unroll
    for (int i = 0; i < 8; ++i) c[i] = bias;

#pragma unroll
    for (int k0 = 0; k0 < HIDDEN; k0 += 32) {
      v16h a;
#pragma unroll
      for (int i = 0; i < 8; ++i) {
        a[i]     = hrow[k0 + kb16 + i];
        a[8 + i] = hrow[k0 + kb16 + 8 + i];
      }
      const v16h b = *(const v16h*)(W2h + (size_t)(nb + m) * HIDDEN + k0 + kb16);
      c = __builtin_amdgcn_wmma_f32_16x16x32_f16(false, a, false, b,
                                                 (short)0, c, false, false);
    }
#pragma unroll
    for (int i = 0; i < 8; ++i)
      orow0[(rbase + i) * HIDDEN + nb + m] = c[i];
  }
}

// ---------------------------------------------------------------------------
extern "C" void kernel_launch(void* const* d_in, const int* in_sizes, int n_in,
                              void* d_out, int out_size, void* d_ws, size_t ws_size,
                              hipStream_t stream) {
  const float* x    = (const float*)d_in[0];
  const float* rbf  = (const float*)d_in[1];
  const int*   ei   = (const int*)  d_in[2];
  const float* Wrbf = (const float*)d_in[3];
  const float* W1   = (const float*)d_in[4];
  const float* b1   = (const float*)d_in[5];
  const float* W2   = (const float*)d_in[6];
  const float* b2   = (const float*)d_in[7];
  float* out = (float*)d_out;

  const int Nn = in_sizes[0] / HIDDEN;   // 50000
  const int E  = in_sizes[1] / NRAD;     // 600000

  // Workspace layout: [acc f32 N*128][W1h f16 128*128][W2h f16 128*128]
  float*    acc      = (float*)d_ws;
  size_t    accElems = (size_t)Nn * HIDDEN;
  _Float16* W1h      = (_Float16*)((char*)d_ws + accElems * sizeof(float));
  _Float16* W2h      = W1h + HIDDEN * HIDDEN;

  {
    int nvec = (int)(accElems / 4);
    zero_acc<<<(nvec + 255) / 256, 256, 0, stream>>>(acc, (int)accElems);
  }
  cvt_w<<<(2 * HIDDEN * HIDDEN + 255) / 256, 256, 0, stream>>>(W1, W2, W1h, W2h,
                                                               HIDDEN * HIDDEN);
  edge_scatter<<<1024, 256, 0, stream>>>(x, rbf, ei, Wrbf, acc, E);

  const int ntiles = Nn / 16;            // 3125, exact
  mlp<<<(ntiles + 7) / 8, 256, 0, stream>>>(acc, W1h, b1, W2h, b2, out, ntiles);
}
